// CrossAttention_1675037245931
// MI455X (gfx1250) — compile-verified, matrix operand order checked by reading, not run on hardware
//
#include <hip/hip_runtime.h>

// ---------------------------------------------------------------------------
// Cross-attention (non-softmax bilinear) for B=2, C=256, C8=32, D=16, H=64, W=64
// One workgroup (256 thr = 8 waves, wave32) per (b,d,h) slice; all GEMMs via
// v_wmma_f32_16x16x32_bf16 with f32 accumulation. Memory-bound: 384 MB stream
// at 23.3 TB/s; bf16 WMMA keeps compute ~3x under the HBM floor.
// ---------------------------------------------------------------------------

typedef __bf16        bf16x16 __attribute__((ext_vector_type(16)));
typedef __bf16        bf16x2  __attribute__((ext_vector_type(2)));
typedef float         f32x8   __attribute__((ext_vector_type(8)));
typedef unsigned int  u32x4   __attribute__((ext_vector_type(4)));

#define CC    256
#define C8    32
#define DHW   65536            // D*H*W = 16*64*64
#define NSLICE 2048            // B*D*H

// LDS pitches (bf16 elements); all multiples of 8 so 16B fragment loads stay aligned
#define XP 264                 // x1T/x2T: [64][256]+pad
#define QP 40                  // qT/kT:   [64][32]+pad
#define AP 72                  // attnS:   [64][64]+pad
#define VP 72                  // vT:      [256][64]+pad

// dynamic-LDS carve (ushort offsets); vT aliases x1T (dead after phase 1)
#define OFF_X2  0              // 64*264 = 16896
#define OFF_Q   16896          // 64*40  =  2560
#define OFF_K   19456          // 64*40  =  2560
#define OFF_A   22016          // 64*72  =  4608
#define OFF_X1  26624          // 64*264 = 16896
#define OFF_V   26624          // 256*72 = 18432 (overlaps X1)
#define SMEM_USHORT 45056
#define SMEM_BYTES  (SMEM_USHORT * 2)   // 90112 B <= 320KB WGP LDS

// Native bf16 conversions (lowers to v_cvt_pk_bf16_f32 on gfx1250; RTNE)
__device__ __forceinline__ unsigned short f2bf(float f) {
  union { __bf16 h; unsigned short u; } cv;
  cv.h = (__bf16)f;
  return cv.u;
}
__device__ __forceinline__ unsigned int f2bf2(float lo, float hi) {
  union { bf16x2 v; unsigned int u; } cv;
  cv.v = bf16x2{(__bf16)lo, (__bf16)hi};
  return cv.u;
}

union Frag {
  bf16x16 v;
  u32x4   q[2];
};

// 16-bit A/B fragment per CDNA5 ISA layout (A and B mirror each other):
// lanes 0-15 : row = row0+lane,      K = kb+0..7  and kb+16..23
// lanes 16-31: row = row0+(lane&15), K = kb+8..15 and kb+24..31
__device__ __forceinline__ Frag load_frag(const unsigned short* base, int row0,
                                          int kb, int pitch, int lane) {
  Frag f;
  const int r   = row0 + (lane & 15);
  const int sel = (lane & 16) ? 8 : 0;
  const u32x4* p = (const u32x4*)(base + r * pitch + kb + sel);
  f.q[0] = p[0];   // K = kb+sel   .. +7
  f.q[1] = p[2];   // K = kb+sel+16.. +23
  return f;
}

__device__ __forceinline__ f32x8 wmma_bf16(const Frag& a, const Frag& b, f32x8 c) {
  return __builtin_amdgcn_wmma_f32_16x16x32_bf16(false, a.v, false, b.v,
                                                 (short)0, c, false, false);
}

// --- pre-pass: fp32 weights -> bf16 in workspace ---------------------------
// ws layout (ushort): Wq [0,8192) | Wk [8192,16384) | Wv [16384,81920)
__global__ __launch_bounds__(256) void cvt_weights(const float* __restrict__ Wq,
                                                   const float* __restrict__ Wk,
                                                   const float* __restrict__ Wv,
                                                   unsigned short* __restrict__ ws) {
  int i = blockIdx.x * 256 + threadIdx.x;
  if (i < 8192)        ws[i] = f2bf(Wq[i]);
  else if (i < 16384)  ws[i] = f2bf(Wk[i - 8192]);
  else if (i < 81920)  ws[i] = f2bf(Wv[i - 16384]);
}

// --- main fused kernel -----------------------------------------------------
__global__ __launch_bounds__(256) void xattn(const float* __restrict__ x1,
                                             const float* __restrict__ x2,
                                             const float* __restrict__ bq,
                                             const float* __restrict__ bk,
                                             const float* __restrict__ bv,
                                             const unsigned short* __restrict__ wbf,
                                             float* __restrict__ out) {
  extern __shared__ unsigned short smem[];
  unsigned short* x2T = smem + OFF_X2;
  unsigned short* qT  = smem + OFF_Q;
  unsigned short* kT  = smem + OFF_K;
  unsigned short* aT  = smem + OFF_A;
  unsigned short* x1T = smem + OFF_X1;
  unsigned short* vT  = smem + OFF_V;   // aliases x1T

  const unsigned short* wqb = wbf;
  const unsigned short* wkb = wbf + 8192;
  const unsigned short* wvb = wbf + 16384;

  const int tid  = threadIdx.x;
  const int lane = tid & 31;
  const int wid  = tid >> 5;                 // wave 0..7
  const int hi   = (lane & 16) ? 8 : 0;      // D-fragment M offset
  const int ln   = lane & 15;                // D-fragment N

  const int bid = blockIdx.x;                // 0..2047
  const int b   = bid >> 10;                 // D*H = 1024
  const int dh  = bid & 1023;
  const int sbase = b * (CC * DHW) + dh * 64;  // + c*DHW + w

  // ---- stage x1/x2 slice -> LDS, transposed, bf16 -------------------------
  // Thread owns fixed y, gathers 8 consecutive channels: lanes sweep
  // consecutive y (coalesced 128B wave requests, channel offsets immediate),
  // then one conflict-free packed ds_store_b128 per array per iteration.
  {
    const int y   = tid & 63;
    const int cb0 = (tid >> 6) * 8;          // 0,8,16,24
    const float* p1 = x1 + sbase + y;
    const float* p2 = x2 + sbase + y;
    #pragma unroll 2
    for (int i = 0; i < 8; ++i) {
      const int c0 = i * 32 + cb0;
      float a[8], e[8];
      #pragma unroll
      for (int j = 0; j < 8; ++j) {
        a[j] = __builtin_nontemporal_load(p1 + (c0 + j) * DHW);
        e[j] = __builtin_nontemporal_load(p2 + (c0 + j) * DHW);
      }
      u32x4 pa, pe;
      pa.x = f2bf2(a[0], a[1]); pa.y = f2bf2(a[2], a[3]);
      pa.z = f2bf2(a[4], a[5]); pa.w = f2bf2(a[6], a[7]);
      pe.x = f2bf2(e[0], e[1]); pe.y = f2bf2(e[2], e[3]);
      pe.z = f2bf2(e[4], e[5]); pe.w = f2bf2(e[6], e[7]);
      *(u32x4*)(x1T + y * XP + c0) = pa;
      *(u32x4*)(x2T + y * XP + c0) = pe;
    }
  }
  __syncthreads();

  // ---- phase 1: q = Wq@x1 + bq, k = Wk@x2 + bk; stored transposed ---------
  // 16 output tiles (q: 2x4, k: 2x4); 2 per wave; K = 256 in 8 steps
  for (int t = 0; t < 2; ++t) {
    const int tile = wid * 2 + t;
    const bool isK = (tile & 8) != 0;
    const unsigned short* wmat = isK ? wkb : wqb;
    const unsigned short* xT   = isK ? x2T : x1T;
    const float*          bias = isK ? bk  : bq;
    unsigned short*       dst  = isK ? kT  : qT;
    const int mt = (tile >> 2) & 1;          // channel tile (C8=32 -> 2)
    const int nt = tile & 3;                 // W tile
    f32x8 acc = {0.f, 0.f, 0.f, 0.f, 0.f, 0.f, 0.f, 0.f};
    #pragma unroll
    for (int kb = 0; kb < 256; kb += 32) {
      Frag a  = load_frag(wmat, mt * 16, kb, CC, lane);  // A: q/k row, K=cin
      Frag bb = load_frag(xT,   nt * 16, kb, XP, lane);  // B: K=cin, N=w
      acc = wmma_bf16(a, bb, acc);
    }
    const int c0 = mt * 16 + hi;
    u32x4 pk;
    pk.x = f2bf2(acc[0] + bias[c0 + 0], acc[1] + bias[c0 + 1]);
    pk.y = f2bf2(acc[2] + bias[c0 + 2], acc[3] + bias[c0 + 3]);
    pk.z = f2bf2(acc[4] + bias[c0 + 4], acc[5] + bias[c0 + 5]);
    pk.w = f2bf2(acc[6] + bias[c0 + 6], acc[7] + bias[c0 + 7]);
    // transposed store: qT[x][c] — 8 consecutive cols => one b128 store
    *(u32x4*)(dst + (nt * 16 + ln) * QP + c0) = pk;
  }
  __syncthreads();

  // ---- phase 2: attn, computed transposed: E[y][x] = k^T q ---------------
  // A = kT (M=y, K=c8), B = qT (K=c8, N=x); D rows = consecutive y per lane
  for (int t = 0; t < 2; ++t) {
    const int tile = wid * 2 + t;            // 0..15 = 4x4 tiles
    const int yt = tile >> 2, xt = tile & 3;
    Frag a  = load_frag(kT, yt * 16, 0, QP, lane);
    Frag bb = load_frag(qT, xt * 16, 0, QP, lane);
    f32x8 acc = {0.f, 0.f, 0.f, 0.f, 0.f, 0.f, 0.f, 0.f};
    acc = wmma_bf16(a, bb, acc);
    u32x4 pk;
    pk.x = f2bf2(acc[0], acc[1]);
    pk.y = f2bf2(acc[2], acc[3]);
    pk.z = f2bf2(acc[4], acc[5]);
    pk.w = f2bf2(acc[6], acc[7]);
    // aT[x][y]: row = x (lane), cols = 8 consecutive y => one b128 store
    *(u32x4*)(aT + (xt * 16 + ln) * AP + yt * 16 + hi) = pk;
  }

  // ---- phase 2.5: v, computed transposed: E[y][c] = x2^T Wv^T + bv -------
  // A = x2T (M=y, K=cin), B = Wv row-major (K=cin, N=cout)
  for (int t = 0; t < 8; ++t) {
    const int tile = wid * 8 + t;            // 0..63 = 4x16 tiles
    const int yt = tile & 3, ct = tile >> 2;
    f32x8 acc = {0.f, 0.f, 0.f, 0.f, 0.f, 0.f, 0.f, 0.f};
    #pragma unroll
    for (int kb = 0; kb < 256; kb += 32) {
      Frag a  = load_frag(x2T, yt * 16, kb, XP, lane);
      Frag bb = load_frag(wvb, ct * 16, kb, CC, lane);
      acc = wmma_bf16(a, bb, acc);
    }
    const float bias = bv[ct * 16 + ln];     // lane-uniform over the 8 rows
    u32x4 pk;
    pk.x = f2bf2(acc[0] + bias, acc[1] + bias);
    pk.y = f2bf2(acc[2] + bias, acc[3] + bias);
    pk.z = f2bf2(acc[4] + bias, acc[5] + bias);
    pk.w = f2bf2(acc[6] + bias, acc[7] + bias);
    // vT[c][y]: row = c (lane), cols = 8 consecutive y => one b128 store
    *(u32x4*)(vT + (ct * 16 + ln) * VP + yt * 16 + hi) = pk;
  }
  __syncthreads();

  // ---- phase 3: out = v x attn^T (M=c, K=y=64, N=x); fp32 NT store --------
  for (int t = 0; t < 8; ++t) {
    const int tile = wid * 8 + t;            // 0..63 = 16x4 tiles
    const int mt = tile >> 2, nt = tile & 3;
    f32x8 acc = {0.f, 0.f, 0.f, 0.f, 0.f, 0.f, 0.f, 0.f};
    #pragma unroll
    for (int kb = 0; kb < 64; kb += 32) {
      Frag a  = load_frag(vT, mt * 16, kb, VP, lane);   // A: v[c][y]
      Frag bb = load_frag(aT, nt * 16, kb, AP, lane);   // B: attn[x][y] rows
      acc = wmma_bf16(a, bb, acc);
    }
    const int x  = nt * 16 + ln;
    const int c0 = mt * 16 + hi;
    #pragma unroll
    for (int r = 0; r < 8; ++r)
      __builtin_nontemporal_store(acc[r], out + sbase + (c0 + r) * DHW + x);
  }
}

extern "C" void kernel_launch(void* const* d_in, const int* in_sizes, int n_in,
                              void* d_out, int out_size, void* d_ws, size_t ws_size,
                              hipStream_t stream) {
  const float* x1 = (const float*)d_in[0];
  const float* x2 = (const float*)d_in[1];
  const float* Wq = (const float*)d_in[2];
  const float* bq = (const float*)d_in[3];
  const float* Wk = (const float*)d_in[4];
  const float* bk = (const float*)d_in[5];
  const float* Wv = (const float*)d_in[6];
  const float* bv = (const float*)d_in[7];
  float* out = (float*)d_out;
  unsigned short* wbf = (unsigned short*)d_ws;   // 81920 ushort = 160 KB

  cvt_weights<<<320, 256, 0, stream>>>(Wq, Wk, Wv, wbf);
  xattn<<<NSLICE, 256, SMEM_BYTES, stream>>>(x1, x2, bq, bk, bv, wbf, out);
}
